// MultiHeadSelfAttention_75642964017394
// MI455X (gfx1250) — matrix operand order repaired and checked by reading, not compiled, not run on hardware
//
#include <hip/hip_runtime.h>
#include <hip/hip_bf16.h>
#include <stdint.h>

// Problem constants (match reference)
#define Bc  4
#define Sc  2048
#define Dc  1024
#define Hc  16
#define HDc 64
#define ATT_SCALE 0.125f   // 1/sqrt(64)

typedef __attribute__((ext_vector_type(16))) _Float16 v16h;
typedef __attribute__((ext_vector_type(8)))  _Float16 v8h;
typedef __attribute__((ext_vector_type(8)))  float    v8f;

typedef __attribute__((ext_vector_type(4))) unsigned int u32x4;
typedef __attribute__((ext_vector_type(8))) int          i32x8;
typedef __attribute__((ext_vector_type(4))) int          i32x4;

#if __has_builtin(__builtin_amdgcn_tensor_load_to_lds)
#define FLASH_TDM 1
#else
#define FLASH_TDM 0
#endif

// Assemble a v16h A/B fragment from two contiguous 8-halfword (16B) LDS runs.
static __device__ inline v16h load_frag2(const _Float16* p0, const _Float16* p1) {
  v8h lo = *(const v8h*)p0;
  v8h hi = *(const v8h*)p1;
  v16h r;
#pragma unroll
  for (int i = 0; i < 8; ++i) { r[i] = lo[i]; r[i + 8] = hi[i]; }
  return r;
}

static __device__ inline v8f wmma_f16(v16h a, v16h b, v8f c) {
  // D = A(16x32 f16) * B(32x16 f16) + C(16x16 f32)
  return __builtin_amdgcn_wmma_f32_16x16x32_f16(
      false, a, false, b, (short)0, c, false, false);
}

#if FLASH_TDM
// TDM 2D tile load: nrows x rowlen elements (2B each), global row stride
// `stride_e` elements, LDS destination padded +16B after every 128B row
// (pad_interval code 4 = 32 DWORDs, pad_amount code 3 = 4 DWORDs) so the
// LDS pitch is 72 halfwords, matching the fragment-load layout.
static __device__ inline void tdm_load_2d(uint32_t lds_byte_addr, const void* gsrc,
                                          uint32_t rowlen_e, uint32_t nrows,
                                          uint32_t stride_e,
                                          uint32_t tdim0, uint32_t tdim1) {
  const uint64_t ga = (uint64_t)(uintptr_t)gsrc;
  u32x4 g0;
  g0[0] = 1u;                                   // count=1 (valid user D#)
  g0[1] = lds_byte_addr;                        // lds_addr
  g0[2] = (uint32_t)ga;                         // global_addr[31:0]
  g0[3] = (uint32_t)((ga >> 32) & 0x01FFFFFFu)  // global_addr[56:32]
          | (2u << 30);                         // type = 2 ("image")
  i32x8 g1;
  g1[0] = (int)((1u << 16)                      // data_size = 1 -> 2 bytes
              | (1u << 20)                      // pad_enable
              | (4u << 22)                      // pad_interval: 32 DWORDs
              | (3u << 25));                    // pad_amount: 4 DWORDs
  g1[1] = (int)((tdim0 & 0xFFFFu) << 16);       // tensor_dim0[15:0]
  g1[2] = (int)((tdim0 >> 16) | ((tdim1 & 0xFFFFu) << 16)); // d0 hi | d1 lo
  g1[3] = (int)((tdim1 >> 16) | (rowlen_e << 16));          // d1 hi | tile_dim0
  g1[4] = (int)(nrows & 0xFFFFu);               // tile_dim1 (tile_dim2 = 0)
  g1[5] = (int)stride_e;                        // tensor_dim0_stride[31:0]
  g1[6] = 0;                                    // stride0 hi | stride1 lo
  g1[7] = 0;
  i32x4 z4; z4[0] = z4[1] = z4[2] = z4[3] = 0;  // groups 2/3 unused (2D)
  i32x8 z8;
#pragma unroll
  for (int i = 0; i < 8; ++i) z8[i] = 0;
  // This toolchain carries the 6-arg builtin (therock-10.0 header form).
  __builtin_amdgcn_tensor_load_to_lds(g0, g1, z4, z4, z8, 0);
}
#endif

// ---------------------------------------------------------------------------
// Kernel 1: QKV projection  out = x @ W + b  (fp32 in, f16 out)
// Block = 128 threads (4 waves). Tile: 64(m) x 64(n), K-step 32.
// ---------------------------------------------------------------------------
__global__ __launch_bounds__(128) void qkv_gemm_kernel(
    const float* __restrict__ X, const float* __restrict__ W,
    const float* __restrict__ bias, _Float16* __restrict__ out, int transposed)
{
  __shared__ _Float16 xs[64][40];   // A tile (64 x 32, pad 40)
  __shared__ _Float16 wsm[32][72];  // B tile (32 x 64, pad 72)

  const int tid  = threadIdx.x;
  const int wave = tid >> 5, lane = tid & 31;
  const int half = lane >> 4, nl = lane & 15;
  const int r0 = blockIdx.x * 64;   // row tile in [0, B*S)
  const int c0 = blockIdx.y * 64;   // col tile in [0, D)

  v8f acc[4];
#pragma unroll
  for (int nt = 0; nt < 4; ++nt)
#pragma unroll
    for (int j = 0; j < 8; ++j) acc[nt][j] = 0.0f;

  for (int k0 = 0; k0 < Dc; k0 += 32) {
    __syncthreads();
    // Stage X tile 64x32 fp32 -> f16
    {
      const int i  = tid >> 1;
      const int ks = (tid & 1) * 16;
      const float* src = X + (size_t)(r0 + i) * Dc + k0 + ks;
      __builtin_prefetch(src + 32, 0, 3);      // next K-step, same row
#pragma unroll
      for (int u = 0; u < 4; ++u) {
        float4 f = *(const float4*)(src + u * 4);
        _Float16* dst = &xs[i][ks + u * 4];
        dst[0] = (_Float16)f.x; dst[1] = (_Float16)f.y;
        dst[2] = (_Float16)f.z; dst[3] = (_Float16)f.w;
      }
    }
    // Stage W tile 32x64 fp32 -> f16
    {
      const int kk = tid >> 2;
      const int cs = (tid & 3) * 16;
      const float* src = W + (size_t)(k0 + kk) * Dc + c0 + cs;
      __builtin_prefetch(src + (size_t)32 * Dc, 0, 3); // next K-step rows
#pragma unroll
      for (int u = 0; u < 4; ++u) {
        float4 f = *(const float4*)(src + u * 4);
        _Float16* dst = &wsm[kk][cs + u * 4];
        dst[0] = (_Float16)f.x; dst[1] = (_Float16)f.y;
        dst[2] = (_Float16)f.z; dst[3] = (_Float16)f.w;
      }
    }
    __syncthreads();

    v16h a = load_frag2(&xs[wave * 16 + nl][half * 8],
                        &xs[wave * 16 + nl][16 + half * 8]);
#pragma unroll
    for (int nt = 0; nt < 4; ++nt) {
      v16h b;
      for (int s = 0; s < 8; ++s) { b[s] = wsm[half * 8 + s][nt * 16 + nl]; b[s + 8] = wsm[16 + half * 8 + s][nt * 16 + nl]; }
      acc[nt] = wmma_f16(a, b, acc[nt]);
    }
  }

  // Epilogue: bias + scatter-store f16 into head-split layout.
  // A 64-row tile never crosses a batch boundary (2048 % 64 == 0).
  const int bb     = r0 >> 11;        // r0 / Sc
  const int s_base = (r0 & (Sc - 1)) + wave * 16 + 8 * half;
#pragma unroll
  for (int nt = 0; nt < 4; ++nt) {
    const int c = c0 + nt * 16 + nl;
    const float bc = bias[c];
    const int h = c >> 6;
    const int d = c & 63;
#pragma unroll
    for (int j = 0; j < 8; ++j) {
      const int ss = s_base + j;
      const float v = acc[nt][j] + bc;
      size_t idx;
      if (transposed) idx = (((size_t)bb * Hc + h) * HDc + d) * Sc + ss;
      else            idx = (((size_t)bb * Hc + h) * Sc + ss) * HDc + d;
      out[idx] = (_Float16)v;
    }
  }
}

// ---------------------------------------------------------------------------
// Kernel 2: fused flash attention. Block = (b, h, 64 q-rows); 4 waves, each
// owning a 16-row q strip. K/V tiles are DMA'd by the Tensor Data Mover into
// double-buffered LDS (pad feature produces the 72-halfword pitch), with the
// next tile's DMA overlapping the current tile's WMMAs.
// ---------------------------------------------------------------------------
__global__ __launch_bounds__(128) void flash_attn_kernel(
    const _Float16* __restrict__ Q, const _Float16* __restrict__ Kt,
    const _Float16* __restrict__ V, _Float16* __restrict__ Ctx)
{
  __shared__ _Float16 q_lds[64][72];         // [q][d]
  __shared__ _Float16 kt_lds[2][64][72];     // [d][k] double-buffered
  __shared__ _Float16 v_lds[2][64][72];      // [k][d] double-buffered
  __shared__ _Float16 p_lds[4][16][72];      // per-wave P re-layout buffer

  const int tid  = threadIdx.x;
  const int wave = tid >> 5, lane = tid & 31;
  const int half = lane >> 4, nl = lane & 15;

  const int nkt = Sc / 64;                   // 32 K/V tiles
  const int qb = blockIdx.x % nkt;
  const int h  = (blockIdx.x / nkt) % Hc;
  const int b  = blockIdx.x / (nkt * Hc);
  const int q0 = qb * 64;

  const size_t head = ((size_t)b * Hc + h) * (size_t)Sc * HDc;

  // Stage Q tile 64x64 f16 (one-time)
  {
    const int i  = tid >> 1;
    const int ds = (tid & 1) * 32;
    const _Float16* src = Q + head + (size_t)(q0 + i) * HDc + ds;
#pragma unroll
    for (int u = 0; u < 4; ++u)
      *(v8h*)&q_lds[i][ds + u * 8] = *(const v8h*)(src + u * 8);
  }

#if FLASH_TDM
  if (wave == 0) {
    tdm_load_2d((uint32_t)(size_t)&kt_lds[0][0][0], Kt + head,
                64, 64, Sc, Sc, HDc);
    tdm_load_2d((uint32_t)(size_t)&v_lds[0][0][0],  V + head,
                64, 64, HDc, HDc, Sc);
    __builtin_amdgcn_s_wait_tensorcnt(0);
  }
#endif
  __syncthreads();

  v16h qa[2];
#pragma unroll
  for (int kk = 0; kk < 2; ++kk)
    qa[kk] = load_frag2(&q_lds[wave * 16 + nl][kk * 32 + half * 8],
                        &q_lds[wave * 16 + nl][kk * 32 + 16 + half * 8]);

  v8f O[4];
  float Mrow[8], Srow[8];
#pragma unroll
  for (int dt = 0; dt < 4; ++dt)
#pragma unroll
    for (int j = 0; j < 8; ++j) O[dt][j] = 0.0f;
#pragma unroll
  for (int j = 0; j < 8; ++j) { Mrow[j] = -1e30f; Srow[j] = 0.0f; }

  for (int it = 0; it < nkt; ++it) {
    const int k0 = it * 64;
#if FLASH_TDM
    const int cur = it & 1, nxt = cur ^ 1;
    // Kick off next tile's DMA; it runs while all waves compute on `cur`.
    if (wave == 0 && it + 1 < nkt) {
      const int kn = k0 + 64;
      tdm_load_2d((uint32_t)(size_t)&kt_lds[nxt][0][0], Kt + head + kn,
                  64, 64, Sc, Sc, HDc);
      tdm_load_2d((uint32_t)(size_t)&v_lds[nxt][0][0],
                  V + head + (size_t)kn * HDc, 64, 64, HDc, HDc, Sc);
    }
#else
    const int cur = 0;
    __syncthreads();
    {
      const int dd = tid >> 1;
      const int ss = (tid & 1) * 32;
      const _Float16* src = Kt + head + (size_t)dd * Sc + k0 + ss;
#pragma unroll
      for (int u = 0; u < 4; ++u)
        *(v8h*)&kt_lds[0][dd][ss + u * 8] = *(const v8h*)(src + u * 8);
    }
    {
      const int kk = tid >> 1;
      const int ds = (tid & 1) * 32;
      const _Float16* src = V + head + (size_t)(k0 + kk) * HDc + ds;
#pragma unroll
      for (int u = 0; u < 4; ++u)
        *(v8h*)&v_lds[0][kk][ds + u * 8] = *(const v8h*)(src + u * 8);
    }
    __syncthreads();
#endif

    // Scores: 16(q) x 64(k) per wave via 8 WMMAs
    v8f sc[4];
#pragma unroll
    for (int nt = 0; nt < 4; ++nt) {
      v8f z;
#pragma unroll
      for (int j = 0; j < 8; ++j) z[j] = 0.0f;
#pragma unroll
      for (int kk = 0; kk < 2; ++kk) {
        v16h bfr;
        for (int s = 0; s < 8; ++s) { bfr[s] = kt_lds[cur][kk * 32 + half * 8 + s][nt * 16 + nl]; bfr[s + 8] = kt_lds[cur][kk * 32 + 16 + half * 8 + s][nt * 16 + nl]; }
        z = wmma_f16(qa[kk], bfr, z);
      }
      sc[nt] = z;
    }

    // Online softmax (C-tile layout: row m = j + 8*half, col = lane%16)
    float tm[8];
#pragma unroll
    for (int j = 0; j < 8; ++j)
      tm[j] = fmaxf(fmaxf(sc[0][j], sc[1][j]), fmaxf(sc[2][j], sc[3][j])) * ATT_SCALE;
#pragma unroll
    for (int off = 1; off < 16; off <<= 1)
#pragma unroll
      for (int j = 0; j < 8; ++j) tm[j] = fmaxf(tm[j], __shfl_xor(tm[j], off, 32));

    float alpha[8], mnew[8], rs[8];
#pragma unroll
    for (int j = 0; j < 8; ++j) {
      mnew[j]  = fmaxf(Mrow[j], tm[j]);
      alpha[j] = __expf(Mrow[j] - mnew[j]);
      Mrow[j]  = mnew[j];
      rs[j]    = 0.0f;
    }
#pragma unroll
    for (int nt = 0; nt < 4; ++nt)
#pragma unroll
      for (int j = 0; j < 8; ++j) {
        float p = __expf(sc[nt][j] * ATT_SCALE - mnew[j]);
        rs[j] += p;
        p_lds[wave][j + 8 * half][nt * 16 + nl] = (_Float16)p;
      }
#pragma unroll
    for (int off = 1; off < 16; off <<= 1)
#pragma unroll
      for (int j = 0; j < 8; ++j) rs[j] += __shfl_xor(rs[j], off, 32);
#pragma unroll
    for (int j = 0; j < 8; ++j) Srow[j] = Srow[j] * alpha[j] + rs[j];
#pragma unroll
    for (int dt = 0; dt < 4; ++dt)
#pragma unroll
      for (int j = 0; j < 8; ++j) O[dt][j] *= alpha[j];

    // Cross-lane LDS dependency is intra-wave (lockstep): one dscnt wait.
    asm volatile("s_wait_dscnt 0" ::: "memory");

    v16h pa[2];
#pragma unroll
    for (int kk = 0; kk < 2; ++kk)
      pa[kk] = load_frag2(&p_lds[wave][nl][kk * 32 + half * 8],
                          &p_lds[wave][nl][kk * 32 + 16 + half * 8]);
#pragma unroll
    for (int dt = 0; dt < 4; ++dt)
#pragma unroll
      for (int kk = 0; kk < 2; ++kk) {
        v16h bfr;
        for (int s = 0; s < 8; ++s) { bfr[s] = v_lds[cur][kk * 32 + half * 8 + s][dt * 16 + nl]; bfr[s + 8] = v_lds[cur][kk * 32 + 16 + half * 8 + s][dt * 16 + nl]; }
        O[dt] = wmma_f16(pa[kk], bfr, O[dt]);
      }

#if FLASH_TDM
    // Next tile's DMA must be complete, and every wave must be done reading
    // `cur`, before the roles swap.
    if (wave == 0 && it + 1 < nkt) __builtin_amdgcn_s_wait_tensorcnt(0);
    __syncthreads();
#endif
  }

  // Finalize: O / rowsum -> ctx f16 in [B*S, D] (heads re-interleaved)
#pragma unroll
  for (int j = 0; j < 8; ++j) Srow[j] = 1.0f / Srow[j];
#pragma unroll
  for (int dt = 0; dt < 4; ++dt) {
    const int c = h * HDc + dt * 16 + nl;
#pragma unroll
    for (int j = 0; j < 8; ++j) {
      const int r = q0 + wave * 16 + j + 8 * half;
      Ctx[((size_t)b * Sc + r) * Dc + c] = (_Float16)(O[dt][j] * Srow[j]);
    }
  }
}

// ---------------------------------------------------------------------------
// Kernel 3: output projection  out = ctx(f16) @ wo + bo  (f32 out)
// ---------------------------------------------------------------------------
__global__ __launch_bounds__(128) void out_proj_kernel(
    const _Float16* __restrict__ A, const float* __restrict__ W,
    const float* __restrict__ bias, float* __restrict__ out)
{
  __shared__ _Float16 as[64][40];
  __shared__ _Float16 wsm[32][72];

  const int tid  = threadIdx.x;
  const int wave = tid >> 5, lane = tid & 31;
  const int half = lane >> 4, nl = lane & 15;
  const int r0 = blockIdx.x * 64;
  const int c0 = blockIdx.y * 64;

  v8f acc[4];
#pragma unroll
  for (int nt = 0; nt < 4; ++nt)
#pragma unroll
    for (int j = 0; j < 8; ++j) acc[nt][j] = 0.0f;

  for (int k0 = 0; k0 < Dc; k0 += 32) {
    __syncthreads();
    {
      const int i  = tid >> 1;
      const int ks = (tid & 1) * 16;
      const _Float16* src = A + (size_t)(r0 + i) * Dc + k0 + ks;
      __builtin_prefetch(src + 32, 0, 3);
      *(v8h*)&as[i][ks]     = *(const v8h*)src;
      *(v8h*)&as[i][ks + 8] = *(const v8h*)(src + 8);
    }
    {
      const int kk = tid >> 2;
      const int cs = (tid & 3) * 16;
      const float* src = W + (size_t)(k0 + kk) * Dc + c0 + cs;
      __builtin_prefetch(src + (size_t)32 * Dc, 0, 3);
#pragma unroll
      for (int u = 0; u < 4; ++u) {
        float4 f = *(const float4*)(src + u * 4);
        _Float16* dst = &wsm[kk][cs + u * 4];
        dst[0] = (_Float16)f.x; dst[1] = (_Float16)f.y;
        dst[2] = (_Float16)f.z; dst[3] = (_Float16)f.w;
      }
    }
    __syncthreads();

    v16h a = load_frag2(&as[wave * 16 + nl][half * 8],
                        &as[wave * 16 + nl][16 + half * 8]);
#pragma unroll
    for (int nt = 0; nt < 4; ++nt) {
      v16h b;
      for (int s = 0; s < 8; ++s) { b[s] = wsm[half * 8 + s][nt * 16 + nl]; b[s + 8] = wsm[16 + half * 8 + s][nt * 16 + nl]; }
      acc[nt] = wmma_f16(a, b, acc[nt]);
    }
  }

#pragma unroll
  for (int nt = 0; nt < 4; ++nt) {
    const int c = c0 + nt * 16 + nl;
    const float bc = bias[c];
#pragma unroll
    for (int j = 0; j < 8; ++j) {
      const int r = r0 + wave * 16 + j + 8 * half;
      out[(size_t)r * Dc + c] = acc[nt][j] + bc;
    }
  }
}

// ---------------------------------------------------------------------------
extern "C" void kernel_launch(void* const* d_in, const int* in_sizes, int n_in,
                              void* d_out, int out_size, void* d_ws, size_t ws_size,
                              hipStream_t stream) {
  (void)in_sizes; (void)n_in; (void)out_size; (void)ws_size;
  const float* x  = (const float*)d_in[0];
  const float* wq = (const float*)d_in[1];
  const float* bq = (const float*)d_in[2];
  const float* wk = (const float*)d_in[3];
  const float* bk = (const float*)d_in[4];
  const float* wv = (const float*)d_in[5];
  const float* bv = (const float*)d_in[6];
  const float* wo = (const float*)d_in[7];
  const float* bo = (const float*)d_in[8];
  float* out = (float*)d_out;

  const size_t per = (size_t)Bc * Hc * Sc * HDc;   // 8 Mi f16 elements
  _Float16* Qh  = (_Float16*)d_ws;
  _Float16* Kt  = Qh + per;
  _Float16* Vh  = Kt + per;
  _Float16* Ctx = Vh + per;

  dim3 blk(128);
  dim3 gmm((Bc * Sc) / 64, Dc / 64);               // 128 x 16 tiles

  qkv_gemm_kernel<<<gmm, blk, 0, stream>>>(x, wq, bq, Qh, 0);
  qkv_gemm_kernel<<<gmm, blk, 0, stream>>>(x, wk, bk, Kt, 1);
  qkv_gemm_kernel<<<gmm, blk, 0, stream>>>(x, wv, bv, Vh, 0);

  flash_attn_kernel<<<dim3(Bc * Hc * (Sc / 64)), blk, 0, stream>>>(Qh, Kt, Vh, Ctx);

  out_proj_kernel<<<gmm, blk, 0, stream>>>(Ctx, wo, bo, out);
}
